// ResnetBlockDDPMpp_Adagn_22127671509642
// MI455X (gfx1250) — compile-verified, matrix operand order checked by reading, not run on hardware
//
#include <hip/hip_runtime.h>
#include <hip/hip_bf16.h>
#include <math.h>

// ---------------- problem constants ----------------
#define BATCH 8
#define CIN   256
#define COUT  256
#define HH    64
#define WW    64
#define HP    66              // halo-padded spatial dim
#define TEMB  512
#define ZEMB  256
#define K2    9
#define KTOT  (K2 * CIN)       // 2304
#define NPIX  (HH * WW)        // 4096

// ---------------- workspace layout (bytes) ----------------
// h0p bf16 NHWC padded [8][66][66][256]  = 17,842,176
// h2p bf16 NHWC padded                  = 17,842,176
// wOff bf16 [32][9][256], wDcn/wC1 bf16 [256][9][256]
// offbuf f32 [8][27][64][64], h1 f32 NHWC [8][4096][256]
static constexpr size_t OFF_H0   = 0;
static constexpr size_t OFF_H2   = 17842176;
static constexpr size_t OFF_WOFF = 35684352;
static constexpr size_t OFF_WDCN = 35831808;
static constexpr size_t OFF_WC1  = 37011456;
static constexpr size_t OFF_OFFB = 38191104;
static constexpr size_t OFF_H1   = 41730048;
static constexpr size_t OFF_S0   = 75284480;
static constexpr size_t OFF_S1   = 75300864;
static constexpr size_t OFF_DEMB = 75317248;
static constexpr size_t OFF_MU0  = 75325440;
static constexpr size_t OFF_RS0  = 75326464;
static constexpr size_t OFF_MU1  = 75327488;
static constexpr size_t OFF_RS1  = 75328512;

// ---------------- CDNA5 WMMA types ----------------
typedef __attribute__((ext_vector_type(16))) __bf16 bf16x16;
typedef __attribute__((ext_vector_type(8)))  __bf16 bf16x8;
typedef __attribute__((ext_vector_type(8)))  float  f32x8;

__device__ __forceinline__ f32x8 zero8() {
    f32x8 v;
#pragma unroll
    for (int i = 0; i < 8; ++i) v[i] = 0.0f;
    return v;
}

__device__ __forceinline__ f32x8 wmma_bf16(bf16x16 a, bf16x16 b, f32x8 c) {
    return __builtin_amdgcn_wmma_f32_16x16x32_bf16(
        false, a, false, b, (short)0, c, false, false);
}

__device__ __forceinline__ float silu_f(float x) { return x / (1.0f + expf(-x)); }
__device__ __forceinline__ float sigm_f(float x) { return 1.0f / (1.0f + expf(-x)); }

// Padded-activation A fragment: yy,xx in [-1,64]; always in-bounds -> no branches.
// ISA 16-bit A layout: lanes 0-15 hold K c0+0..7 / c0+16..23, lanes 16-31 c0+8..15 / c0+24..31.
__device__ __forceinline__ bf16x16 load_a_pad(const __bf16* __restrict__ act,
                                              int b, int yy, int xx,
                                              int c0, int koff) {
    const __bf16* p = act + (((size_t)(b * HP + (yy + 1)) * HP + (xx + 1)) << 8)
                          + c0 + koff;
    bf16x8 lo = *(const bf16x8*)p;
    bf16x8 hi = *(const bf16x8*)(p + 16);
    bf16x16 a;
#pragma unroll
    for (int i = 0; i < 8; ++i) { a[i] = lo[i]; a[i + 8] = hi[i]; }
    return a;
}

// B fragment from global weights [N][tap][C]
__device__ __forceinline__ bf16x16 load_b_frag(const __bf16* __restrict__ w,
                                               int n, int tap, int c0, int koff) {
    const __bf16* p = w + (((size_t)n * K2 + tap) << 8) + c0 + koff;
    bf16x8 lo = *(const bf16x8*)p;
    bf16x8 hi = *(const bf16x8*)(p + 16);
    bf16x16 r;
#pragma unroll
    for (int i = 0; i < 8; ++i) { r[i] = lo[i]; r[i + 8] = hi[i]; }
    return r;
}

// B fragment from LDS-staged weight tile [256][256] bf16
__device__ __forceinline__ bf16x16 load_b_lds(const __bf16* wlds,
                                              int n, int c0, int koff) {
    const __bf16* p = wlds + ((n << 8) + c0 + koff);
    bf16x8 lo = *(const bf16x8*)p;
    bf16x8 hi = *(const bf16x8*)(p + 16);
    bf16x16 r;
#pragma unroll
    for (int i = 0; i < 8; ++i) { r[i] = lo[i]; r[i + 8] = hi[i]; }
    return r;
}

// Async global->LDS copy of one 3x3 tap's weight slice: 256 N x 256 C bf16 = 128 KiB.
// Each of 256 threads issues 32 global_load_async_to_lds_b128 (ASYNCcnt tracked).
__device__ __forceinline__ void stage_weights_tap(const __bf16* __restrict__ wsrc,
                                                  int tap, __bf16* wlds, int tid) {
    uint32_t lds_base = (uint32_t)(size_t)(&wlds[0]);
#pragma unroll 4
    for (int k = 0; k < 32; ++k) {
        int q  = tid + (k << 8);            // b128 chunk id, 0..8191
        int n  = q >> 5;                    // output channel
        int qc = q & 31;                    // 16B chunk within channel row
        uint32_t voff = (uint32_t)(n * (K2 * CIN * 2) + tap * (CIN * 2) + (qc << 4));
        uint32_t lds  = lds_base + ((uint32_t)q << 4);
        asm volatile("global_load_async_to_lds_b128 %0, %1, %2"
                     :: "v"(lds), "v"(voff), "s"(wsrc) : "memory");
    }
    asm volatile("s_wait_asynccnt 0x0" ::: "memory");
}

// ---------------- kernel 0: zero padded activation buffers ----------------
__global__ void __launch_bounds__(256)
zero_pad_kernel(uint4* __restrict__ p, int n16) {
    int i = blockIdx.x * 256 + threadIdx.x;
    if (i < n16) { uint4 z = {0u, 0u, 0u, 0u}; p[i] = z; }
}

// ---------------- kernel 1: weight transpose + f32 -> bf16 ----------------
__global__ void __launch_bounds__(256)
prep_weights_kernel(const float* __restrict__ offw, const float* __restrict__ dcnw,
                    const float* __restrict__ c1w,
                    __bf16* __restrict__ wOff, __bf16* __restrict__ wDcn,
                    __bf16* __restrict__ wC1) {
    int idx = blockIdx.x * 256 + threadIdx.x;
    const int SZ_OFF = 32 * KTOT;
    const int SZ_BIG = COUT * KTOT;
    if (idx < SZ_OFF) {
        int n = idx / KTOT, r = idx % KTOT, tap = r >> 8, c = r & 255;
        float v = (n < 27) ? offw[((size_t)(n * CIN + c)) * K2 + tap] : 0.0f;
        wOff[idx] = (__bf16)v;
    } else if (idx < SZ_OFF + SZ_BIG) {
        int i = idx - SZ_OFF;
        int n = i / KTOT, r = i % KTOT, tap = r >> 8, c = r & 255;
        wDcn[i] = (__bf16)dcnw[((size_t)(n * CIN + c)) * K2 + tap];
    } else if (idx < SZ_OFF + 2 * SZ_BIG) {
        int i = idx - SZ_OFF - SZ_BIG;
        int n = i / KTOT, r = i % KTOT, tap = r >> 8, c = r & 255;
        wC1[i] = (__bf16)c1w[((size_t)(n * CIN + c)) * K2 + tap];
    }
}

// ---------------- kernel 2: small dense layers ----------------
__global__ void __launch_bounds__(256)
small_dense_kernel(const float* __restrict__ temb, const float* __restrict__ zemb,
                   const float* __restrict__ g0w, const float* __restrict__ g0b,
                   const float* __restrict__ d0w, const float* __restrict__ d0b,
                   const float* __restrict__ g1w, const float* __restrict__ g1b,
                   float* __restrict__ s0, float* __restrict__ s1,
                   float* __restrict__ demb) {
    int idx = blockIdx.x * 256 + threadIdx.x;
    if (idx >= BATCH * 1280) return;
    int b = idx / 1280, r = idx % 1280;
    if (r < 512) {
        const float* z = zemb + (size_t)b * ZEMB;
        const float* w = g0w + (size_t)r * ZEMB;
        float s = g0b[r];
        for (int k = 0; k < ZEMB; ++k) s += z[k] * w[k];
        s0[b * 512 + r] = s;
    } else if (r < 1024) {
        int j = r - 512;
        const float* z = zemb + (size_t)b * ZEMB;
        const float* w = g1w + (size_t)j * ZEMB;
        float s = g1b[j];
        for (int k = 0; k < ZEMB; ++k) s += z[k] * w[k];
        s1[b * 512 + j] = s;
    } else {
        int j = r - 1024;
        const float* t = temb + (size_t)b * TEMB;
        const float* w = d0w + (size_t)j * TEMB;
        float s = d0b[j];
        for (int k = 0; k < TEMB; ++k) s += silu_f(t[k]) * w[k];
        demb[b * COUT + j] = s;
    }
}

// ---------------- kernel 3: GN stats on x (NCHW) ----------------
__global__ void __launch_bounds__(256)
gn0_stats_kernel(const float* __restrict__ x, float* __restrict__ mu,
                 float* __restrict__ rs) {
    int bg = blockIdx.x, b = bg >> 5, g = bg & 31, t = threadIdx.x;
    float s = 0.f, q = 0.f;
    for (int c = 0; c < 8; ++c) {
        const float* p = x + (((size_t)(b * CIN + g * 8 + c)) << 12);
        for (int i = t; i < NPIX; i += 256) { float v = p[i]; s += v; q += v * v; }
    }
    __shared__ float sh[256], sh2[256];
    sh[t] = s; sh2[t] = q; __syncthreads();
    for (int o = 128; o > 0; o >>= 1) {
        if (t < o) { sh[t] += sh[t + o]; sh2[t] += sh2[t + o]; }
        __syncthreads();
    }
    if (t == 0) {
        float m = sh[0] / 32768.0f;
        float v = sh2[0] / 32768.0f - m * m;
        mu[bg] = m; rs[bg] = rsqrtf(v + 1e-6f);
    }
}

// ---------------- kernel 4: h0p = silu(adagn0(x)) -> bf16 padded NHWC --------
__global__ void __launch_bounds__(256)
act0_kernel(const float* __restrict__ x, const float* __restrict__ s0,
            const float* __restrict__ mu, const float* __restrict__ rs,
            __bf16* __restrict__ h0p) {
    size_t idx = (size_t)blockIdx.x * 256 + threadIdx.x;   // NCHW order
    int b = (int)(idx >> 20);
    int rem = (int)(idx & 1048575);
    int c = rem >> 12, p = rem & 4095;
    int y = p >> 6, xp = p & 63;
    float v = x[idx];
    int bg = (b << 5) + (c >> 3);
    float nv = (v - mu[bg]) * rs[bg];
    float hv = s0[b * 512 + c] * nv + s0[b * 512 + 256 + c];
    h0p[(((size_t)(b * HP + y + 1) * HP + xp + 1) << 8) + c] = (__bf16)silu_f(hv);
}

// ---------------- kernel 5: offset conv, implicit GEMM (N padded to 32) ------
__global__ void __launch_bounds__(256)
conv_offset_kernel(const __bf16* __restrict__ h0p, const __bf16* __restrict__ woff,
                   const float* __restrict__ offset_b, float* __restrict__ offbuf) {
    int tid = threadIdx.x, wave = tid >> 5, lane = tid & 31;
    int mt = blockIdx.x * 4 + (wave >> 1);
    int n0 = (wave & 1) << 4;
    int b = mt >> 8, pt = mt & 255, y = pt >> 2, x0 = (pt & 3) << 4;
    int l15 = lane & 15, hi = (lane >> 4) & 1, koff = hi << 3;
    int xa = x0 + l15;
    f32x8 acc = zero8();
    for (int tap = 0; tap < K2; ++tap) {
        int yy = y + tap / 3 - 1;
        int xx = xa + tap % 3 - 1;
        if (tap < K2 - 1)
            __builtin_prefetch(woff + (((size_t)(n0 + l15) * K2 + tap + 1) << 8), 0, 1);
#pragma unroll
        for (int cc = 0; cc < 8; ++cc) {
            int c0 = cc << 5;
            bf16x16 a  = load_a_pad(h0p, b, yy, xx, c0, koff);
            bf16x16 bb = load_b_frag(woff, n0 + l15, tap, c0, koff);
            acc = wmma_bf16(a, bb, acc);
        }
    }
    int ch = n0 + l15;
    if (ch < 27) {
        float bias = offset_b[ch];
#pragma unroll
        for (int r = 0; r < 8; ++r) {
            int m = r + (hi << 3);
            int p = y * WW + x0 + m;
            offbuf[(((size_t)(b * 27 + ch)) << 12) + p] = acc[r] + bias;
        }
    }
}

// ---------------- kernel 6: deformable conv, fused sample + GEMM -------------
__global__ void __launch_bounds__(256)
dcn_kernel(const __bf16* __restrict__ h0p, const __bf16* __restrict__ wdcn,
           const float* __restrict__ offbuf, const float* __restrict__ dcn_b,
           const float* __restrict__ demb, float* __restrict__ h1) {
    __shared__ __bf16 wlds[COUT * CIN];   // 128 KiB weight tile for current tap
    int tid = threadIdx.x, wave = tid >> 5, lane = tid & 31;
    int mt = blockIdx.x * 2 + (wave >> 2);
    int n0 = (wave & 3) << 6;
    int b = mt >> 8, pt = mt & 255, y = pt >> 2, x0 = (pt & 3) << 4;
    int l15 = lane & 15, hi = (lane >> 4) & 1, koff = hi << 3;
    int xa = x0 + l15;
    f32x8 a0 = zero8(), a1 = zero8(), a2 = zero8(), a3 = zero8();
    const float* ob = offbuf + ((size_t)b * 27 << 12) + y * WW + xa;
    for (int tap = 0; tap < K2; ++tap) {
        __syncthreads();                       // previous tap reads complete
        stage_weights_tap(wdcn, tap, wlds, tid);
        // bilinear sampling parameters for this lane's pixel
        float dy = ob[((size_t)(2 * tap)) << 12];
        float dx = ob[((size_t)(2 * tap + 1)) << 12];
        float mask = sigm_f(ob[((size_t)(18 + tap)) << 12]);
        int ky = tap / 3, kx = tap % 3;
        float py = dy + (float)(ky + y - 1);
        float px = dx + (float)(kx + xa - 1);
        float fy = floorf(py), fx = floorf(px);
        float wy1 = py - fy, wx1 = px - fx;
        int iy = (int)fy, ix = (int)fx;
        const __bf16* cp[4];
        float cw[4];
#pragma unroll
        for (int s = 0; s < 4; ++s) {
            int sy = s >> 1, sx = s & 1;
            int yy = iy + sy, xx = ix + sx;
            bool valid = ((unsigned)yy < (unsigned)HH) && ((unsigned)xx < (unsigned)WW);
            float w = (sy ? wy1 : 1.0f - wy1) * (sx ? wx1 : 1.0f - wx1);
            cw[s] = valid ? w * mask : 0.0f;
            int yc = min(max(yy, 0), HH - 1), xc = min(max(xx, 0), WW - 1);
            cp[s] = h0p + (((size_t)(b * HP + yc + 1) * HP + xc + 1) << 8);
        }
        __syncthreads();                       // weight tile staged for all waves
#pragma unroll
        for (int cc = 0; cc < 8; ++cc) {
            int c0 = cc << 5;
            float af[16];
#pragma unroll
            for (int i = 0; i < 16; ++i) af[i] = 0.0f;
#pragma unroll
            for (int s = 0; s < 4; ++s) {
                bf16x8 v1 = *(const bf16x8*)(cp[s] + c0 + koff);
                bf16x8 v2 = *(const bf16x8*)(cp[s] + c0 + 16 + koff);
                float w = cw[s];
#pragma unroll
                for (int i = 0; i < 8; ++i) {
                    af[i]     += w * (float)v1[i];
                    af[i + 8] += w * (float)v2[i];
                }
            }
            bf16x16 a;
#pragma unroll
            for (int i = 0; i < 16; ++i) a[i] = (__bf16)af[i];
            a0 = wmma_bf16(a, load_b_lds(wlds, n0 + l15,      c0, koff), a0);
            a1 = wmma_bf16(a, load_b_lds(wlds, n0 + 16 + l15, c0, koff), a1);
            a2 = wmma_bf16(a, load_b_lds(wlds, n0 + 32 + l15, c0, koff), a2);
            a3 = wmma_bf16(a, load_b_lds(wlds, n0 + 48 + l15, c0, koff), a3);
        }
    }
#pragma unroll
    for (int j = 0; j < 4; ++j) {
        f32x8 acc = (j == 0) ? a0 : (j == 1) ? a1 : (j == 2) ? a2 : a3;
        int ch = n0 + (j << 4) + l15;
        float bias = dcn_b[ch] + demb[b * COUT + ch];
#pragma unroll
        for (int r = 0; r < 8; ++r) {
            int m = r + (hi << 3);
            int p = y * WW + x0 + m;
            h1[((((size_t)b << 12) + p) << 8) + ch] = acc[r] + bias;
        }
    }
}

// ---------------- kernel 7: GN stats on h1 (NHWC) ----------------
__global__ void __launch_bounds__(256)
gn1_stats_kernel(const float* __restrict__ h1, float* __restrict__ mu,
                 float* __restrict__ rs) {
    int bg = blockIdx.x, b = bg >> 5, g = bg & 31, t = threadIdx.x;
    float s = 0.f, q = 0.f;
    for (int e = t; e < 32768; e += 256) {
        int p = e >> 3, c = (g << 3) + (e & 7);
        float v = h1[((((size_t)b << 12) + p) << 8) + c];
        s += v; q += v * v;
    }
    __shared__ float sh[256], sh2[256];
    sh[t] = s; sh2[t] = q; __syncthreads();
    for (int o = 128; o > 0; o >>= 1) {
        if (t < o) { sh[t] += sh[t + o]; sh2[t] += sh2[t + o]; }
        __syncthreads();
    }
    if (t == 0) {
        float m = sh[0] / 32768.0f;
        float v = sh2[0] / 32768.0f - m * m;
        mu[bg] = m; rs[bg] = rsqrtf(v + 1e-6f);
    }
}

// ---------------- kernel 8: h2p = silu(adagn1(h1)) -> bf16 padded NHWC -------
__global__ void __launch_bounds__(256)
act1_kernel(const float* __restrict__ h1, const float* __restrict__ s1,
            const float* __restrict__ mu, const float* __restrict__ rs,
            __bf16* __restrict__ h2p) {
    size_t idx = (size_t)blockIdx.x * 256 + threadIdx.x;   // NHWC order
    int b = (int)(idx >> 20);
    int rem = (int)(idx & 1048575);
    int p = rem >> 8, c = rem & 255;
    int y = p >> 6, xp = p & 63;
    float v = h1[idx];
    int bg = (b << 5) + (c >> 3);
    float nv = (v - mu[bg]) * rs[bg];
    float hv = s1[b * 512 + c] * nv + s1[b * 512 + 256 + c];
    h2p[(((size_t)(b * HP + y + 1) * HP + xp + 1) << 8) + c] = (__bf16)silu_f(hv);
}

// ---------------- kernel 9: conv1 implicit GEMM + residual ----------------
__global__ void __launch_bounds__(256)
conv1_kernel(const __bf16* __restrict__ h2p, const __bf16* __restrict__ wc1,
             const float* __restrict__ conv1_b, const float* __restrict__ xin,
             float* __restrict__ out) {
    __shared__ __bf16 wlds[COUT * CIN];   // 128 KiB weight tile for current tap
    int tid = threadIdx.x, wave = tid >> 5, lane = tid & 31;
    int mt = blockIdx.x * 2 + (wave >> 2);
    int n0 = (wave & 3) << 6;
    int b = mt >> 8, pt = mt & 255, y = pt >> 2, x0 = (pt & 3) << 4;
    int l15 = lane & 15, hi = (lane >> 4) & 1, koff = hi << 3;
    int xa = x0 + l15;
    f32x8 a0 = zero8(), a1 = zero8(), a2 = zero8(), a3 = zero8();
    for (int tap = 0; tap < K2; ++tap) {
        __syncthreads();
        stage_weights_tap(wc1, tap, wlds, tid);
        __syncthreads();
        int yy = y + tap / 3 - 1;
        int xx = xa + tap % 3 - 1;
#pragma unroll
        for (int cc = 0; cc < 8; ++cc) {
            int c0 = cc << 5;
            bf16x16 a = load_a_pad(h2p, b, yy, xx, c0, koff);
            a0 = wmma_bf16(a, load_b_lds(wlds, n0 + l15,      c0, koff), a0);
            a1 = wmma_bf16(a, load_b_lds(wlds, n0 + 16 + l15, c0, koff), a1);
            a2 = wmma_bf16(a, load_b_lds(wlds, n0 + 32 + l15, c0, koff), a2);
            a3 = wmma_bf16(a, load_b_lds(wlds, n0 + 48 + l15, c0, koff), a3);
        }
    }
#pragma unroll
    for (int j = 0; j < 4; ++j) {
        f32x8 acc = (j == 0) ? a0 : (j == 1) ? a1 : (j == 2) ? a2 : a3;
        int ch = n0 + (j << 4) + l15;
        float bias = conv1_b[ch];
#pragma unroll
        for (int r = 0; r < 8; ++r) {
            int m = r + (hi << 3);
            int p = y * WW + x0 + m;
            size_t o = (((size_t)(b * COUT + ch)) << 12) + p;   // NCHW
            out[o] = acc[r] + bias + xin[o];
        }
    }
}

// ---------------- host launcher ----------------
extern "C" void kernel_launch(void* const* d_in, const int* in_sizes, int n_in,
                              void* d_out, int out_size, void* d_ws, size_t ws_size,
                              hipStream_t stream) {
    (void)in_sizes; (void)n_in; (void)out_size; (void)ws_size;
    const float* x       = (const float*)d_in[0];
    const float* temb    = (const float*)d_in[1];
    const float* zemb    = (const float*)d_in[2];
    const float* g0w     = (const float*)d_in[3];
    const float* g0b     = (const float*)d_in[4];
    const float* offw    = (const float*)d_in[5];
    const float* offb_in = (const float*)d_in[6];
    const float* dcnw    = (const float*)d_in[7];
    const float* dcnb    = (const float*)d_in[8];
    const float* d0w     = (const float*)d_in[9];
    const float* d0b     = (const float*)d_in[10];
    const float* g1w     = (const float*)d_in[11];
    const float* g1b     = (const float*)d_in[12];
    const float* c1w     = (const float*)d_in[13];
    const float* c1b     = (const float*)d_in[14];
    float* out = (float*)d_out;

    char* ws = (char*)d_ws;
    __bf16* h0p  = (__bf16*)(ws + OFF_H0);
    __bf16* h2p  = (__bf16*)(ws + OFF_H2);
    __bf16* wOff = (__bf16*)(ws + OFF_WOFF);
    __bf16* wDcn = (__bf16*)(ws + OFF_WDCN);
    __bf16* wC1  = (__bf16*)(ws + OFF_WC1);
    float* offbuf = (float*)(ws + OFF_OFFB);
    float* h1     = (float*)(ws + OFF_H1);
    float* s0     = (float*)(ws + OFF_S0);
    float* s1     = (float*)(ws + OFF_S1);
    float* demb   = (float*)(ws + OFF_DEMB);
    float* mu0    = (float*)(ws + OFF_MU0);
    float* rs0    = (float*)(ws + OFF_RS0);
    float* mu1    = (float*)(ws + OFF_MU1);
    float* rs1    = (float*)(ws + OFF_RS1);

    // zero both padded activation buffers (35,684,352 B = 2,230,272 uint4)
    zero_pad_kernel<<<8712, 256, 0, stream>>>((uint4*)(ws + OFF_H0), 2230272);

    prep_weights_kernel<<<(1253376 + 255) / 256, 256, 0, stream>>>(
        offw, dcnw, c1w, wOff, wDcn, wC1);

    small_dense_kernel<<<(BATCH * 1280 + 255) / 256, 256, 0, stream>>>(
        temb, zemb, g0w, g0b, d0w, d0b, g1w, g1b, s0, s1, demb);

    gn0_stats_kernel<<<BATCH * 32, 256, 0, stream>>>(x, mu0, rs0);

    act0_kernel<<<(BATCH * CIN * NPIX) / 256, 256, 0, stream>>>(x, s0, mu0, rs0, h0p);

    conv_offset_kernel<<<512, 256, 0, stream>>>(h0p, wOff, offb_in, offbuf);

    dcn_kernel<<<1024, 256, 0, stream>>>(h0p, wDcn, offbuf, dcnb, demb, h1);

    gn1_stats_kernel<<<BATCH * 32, 256, 0, stream>>>(h1, mu1, rs1);

    act1_kernel<<<(BATCH * COUT * NPIX) / 256, 256, 0, stream>>>(h1, s1, mu1, rs1, h2p);

    conv1_kernel<<<1024, 256, 0, stream>>>(h2p, wC1, c1b, x, out);
}
